// RecognitionODERNN_15547781611917
// MI455X (gfx1250) — matrix operand order repaired and checked by reading, not compiled, not run on hardware
//
#include <hip/hip_runtime.h>

// ODE-RNN (backward scan) for MI455X / gfx1250, wave32 + WMMA bf16.
// Batch rows are independent -> 16 workgroups x 16 rows, no inter-WG sync.
// Weights pre-swizzled to bf16 WMMA B-tile layout; state/accum in f32 regs.
// Round 3: zero-C WMMA init (inline SRC2=0, no bias splat), biases staged in
// LDS (frees 12 loop-invariant VGPRs -> no spills), float4 observation loads.

typedef __attribute__((ext_vector_type(16))) __bf16 v16bf;
typedef __attribute__((ext_vector_type(8)))  float  v8f;

constexpr int Bsz  = 256;
constexpr int Tlen = 96;
constexpr int OBS  = 64;
constexpr int LAT  = 256;
constexpr int HID  = 512;

__device__ __forceinline__ unsigned short f2bf(float f) {
  unsigned int u = __float_as_uint(f);
  unsigned int r = u + 0x7FFFu + ((u >> 16) & 1u);   // round-to-nearest-even
  return (unsigned short)(r >> 16);
}

// tanh(x) = 1 - 2/(exp(2x)+1); v_exp_f32 + v_rcp_f32, saturates correctly.
__device__ __forceinline__ float fast_tanh(float x) {
  float e2 = __expf(2.0f * x);
  return 1.0f - 2.0f * __builtin_amdgcn_rcpf(e2 + 1.0f);
}

// ---------------------------------------------------------------------------
// Weight pre-swizzle: row-major f32 [K,N] -> bf16 tiles [kt][nt][512 shorts],
// tile interior = lane*16 + s, lane = n_local + 16*((k_local/8)&1),
// s = k_local%8 + 8*(k_local/16): 16-bit WMMA B layout (same K order as A).
// ---------------------------------------------------------------------------
__global__ void pack_weight(const float* __restrict__ W,
                            unsigned short* __restrict__ dst, int K, int N) {
  int idx = blockIdx.x * blockDim.x + threadIdx.x;
  if (idx >= K * N) return;
  int n = idx % N, k = idx / N;
  int kt = k >> 5, kl = k & 31, nt = n >> 4, nlo = n & 15;
  int NT = N >> 4;
  dst[((kt * NT + nt) << 9) + ((nlo + (((kl >> 3) & 1) << 4)) << 4) + (kl & 7) + ((kl >> 4) << 3)] = f2bf(W[idx]);
}

// A-operand packed-layout (16-bit A 16x32 tile):
//  element (m,k): lane = m + 16*((k/8)&1), vgpr = (k%8)/2 + 4*(k/16), half=k%2.
__device__ __forceinline__ int a_elem_idx(int m, int kglobal) {
  int kt = kglobal >> 5, kl = kglobal & 31;
  int lane_a = m + (((kl >> 3) & 1) << 4);
  return (kt << 9) + (lane_a << 4) + (((kl & 7) >> 1) << 1) + ((kl >> 4) << 3) + (kl & 1);
}

// GEMM1: [16 x 32*KT] (A from aP) x W[32*KT x 512], +bias, tanh, scatter into
// hP in A-layout for GEMM2. Wave w owns N-tiles 4w..4w+3, two passes of 2
// tiles -> only 16 accumulator VGPRs live. Chains start with inline C=0.
template <int KT>
__device__ __forceinline__ void gemm_hid(const unsigned short* aP,
                                         unsigned short* hP,
                                         const unsigned short* __restrict__ Wt,
                                         const float* bL, int w, int lane) {
  const int nl = lane & 15;
  const int hoff = (lane >> 4) << 3;                  // D: M = e + hoff
#pragma unroll
  for (int p = 0; p < 2; ++p) {
    v8f acc[2];
#pragma unroll
    for (int kt = 0; kt < KT; ++kt) {
      const v16bf a = *reinterpret_cast<const v16bf*>(aP + (kt << 9) + (lane << 4));
#pragma unroll
      for (int i = 0; i < 2; ++i) {
        const int nt = w * 4 + p * 2 + i;             // NT_total = 32
        const v16bf b = *reinterpret_cast<const v16bf*>(Wt + (((kt << 5) + nt) << 9) + (lane << 4));
        if (kt == 0) {
          const v8f z = {};
          acc[i] = __builtin_amdgcn_wmma_f32_16x16x32_bf16(false, a, false, b,
                                                           (short)0, z, false, false);
        } else {
          acc[i] = __builtin_amdgcn_wmma_f32_16x16x32_bf16(false, a, false, b,
                                                           (short)0, acc[i], false, false);
        }
      }
    }
#pragma unroll
    for (int i = 0; i < 2; ++i) {
      const int n = (w * 4 + p * 2 + i) * 16 + nl;    // column -> next K index
      const float bias = bL[n];
      const int base = a_elem_idx(0, n);
#pragma unroll
      for (int e = 0; e < 8; ++e)
        hP[base + ((e + hoff) << 4)] = f2bf(fast_tanh(acc[i][e] + bias));
    }
  }
}

// GEMM2: [16 x 32*KT] (A from hP) x W[32*KT x 256] -> D + bias in registers.
// Wave w owns N-tiles 2w, 2w+1 (two interleaved chains for XDL ILP).
template <int KT>
__device__ __forceinline__ void gemm_acc(const unsigned short* hP,
                                         const unsigned short* __restrict__ Wt,
                                         const float* bL, int w, int lane,
                                         v8f (&kv)[2]) {
  const int nl = lane & 15;
  v8f acc[2];
#pragma unroll
  for (int kt = 0; kt < KT; ++kt) {
    const v16bf a = *reinterpret_cast<const v16bf*>(hP + (kt << 9) + (lane << 4));
#pragma unroll
    for (int j = 0; j < 2; ++j) {
      const int nt = w * 2 + j;
      const v16bf b = *reinterpret_cast<const v16bf*>(Wt + (((kt << 4) + nt) << 9) + (lane << 4));
      if (kt == 0) {
        const v8f z = {};
        acc[j] = __builtin_amdgcn_wmma_f32_16x16x32_bf16(false, a, false, b,
                                                         (short)0, z, false, false);
      } else {
        acc[j] = __builtin_amdgcn_wmma_f32_16x16x32_bf16(false, a, false, b,
                                                         (short)0, acc[j], false, false);
      }
    }
  }
#pragma unroll
  for (int j = 0; j < 2; ++j) {
    const float bias = bL[(w * 2 + j) * 16 + nl];
#pragma unroll
    for (int e = 0; e < 8; ++e) kv[j][e] = acc[j][e] + bias;
  }
}

// Scatter thread-owned state elements (m = e+hoff, n = (2w+j)*16+nl) into
// aPack tiles 0..7 as bf16 (A operand for the next latent_ode GEMM1).
__device__ __forceinline__ void pack_state(unsigned short* aP,
                                           const float (&s)[2][8], int w, int lane) {
  const int nl = lane & 15;
  const int hoff = (lane >> 4) << 3;
#pragma unroll
  for (int j = 0; j < 2; ++j) {
    const int n = (w * 2 + j) * 16 + nl;
    const int base = a_elem_idx(0, n);
#pragma unroll
    for (int e = 0; e < 8; ++e)
      aP[base + ((e + hoff) << 4)] = f2bf(s[j][e]);
  }
}

// Load x_t (16 rows x 64 obs) as one float4 per thread, pack into tiles 8,9.
__device__ __forceinline__ void pack_x(unsigned short* aP,
                                       const float* __restrict__ dataset,
                                       int b0, int t, int tid) {
  const int e4 = tid * 4;                             // 256 threads * 4 = 1024
  const int m = e4 >> 6, o = e4 & 63;
  const float4 v = *reinterpret_cast<const float4*>(
      &dataset[((b0 + m) * Tlen + t) * OBS + o]);
  const float vv[4] = {v.x, v.y, v.z, v.w};
#pragma unroll
  for (int u = 0; u < 4; ++u)
    aP[a_elem_idx(m, LAT + o + u)] = f2bf(vv[u]);
}

__global__ __launch_bounds__(256, 1)
__attribute__((amdgpu_waves_per_eu(1, 2)))
void odernn_main(
    const float* __restrict__ dataset, const float* __restrict__ ts,
    const float* __restrict__ b1, const float* __restrict__ b2,
    const float* __restrict__ br1, const float* __restrict__ br2,
    const unsigned short* __restrict__ W1s, const unsigned short* __restrict__ W2s,
    const unsigned short* __restrict__ Wr1s, const unsigned short* __restrict__ Wr2s,
    float* __restrict__ out) {
  __shared__ __align__(32) unsigned short aPack[10 * 512];  // up to 10 K-tiles (rnn)
  __shared__ __align__(32) unsigned short hPack[16 * 512];  // hidden (512) A-tiles
  __shared__ float hStep[16];
  __shared__ float b1L[HID], br1L[HID], b2L[LAT], br2L[LAT];

  const int tid = threadIdx.x;
  const int w = tid >> 5, lane = tid & 31;
  const int b0 = blockIdx.x * 16;
  const int nl = lane & 15;
  const int hoff = (lane >> 4) << 3;

  // Stage biases into LDS once (read-only afterwards).
  for (int i = tid; i < HID; i += 256) { b1L[i] = b1[i]; br1L[i] = br1[i]; }
  for (int i = tid; i < LAT; i += 256) { b2L[i] = b2[i]; br2L[i] = br2[i]; }

  float s_own[2][8];
#pragma unroll
  for (int j = 0; j < 2; ++j)
#pragma unroll
    for (int e = 0; e < 8; ++e) s_own[j][e] = 0.0f;

  // ---- initial RNN update: state += rnn(concat(0, x[T-1])) ----
  pack_state(aPack, s_own, w, lane);
  pack_x(aPack, dataset, b0, Tlen - 1, tid);
  __syncthreads();
  gemm_hid<10>(aPack, hPack, Wr1s, br1L, w, lane);
  __syncthreads();
  {
    v8f kv[2];
    gemm_acc<16>(hPack, Wr2s, br2L, w, lane, kv);
#pragma unroll
    for (int j = 0; j < 2; ++j)
#pragma unroll
      for (int e = 0; e < 8; ++e) s_own[j][e] += kv[j][e];
    pack_state(aPack, s_own, w, lane);
  }
  __syncthreads();

  // ---- backward scan over t = T-2 .. 0 ----
#pragma unroll 1
  for (int t = Tlen - 2; t >= 0; --t) {
    if (tid < 16)
      hStep[tid] = (ts[(b0 + tid) * Tlen + t] - ts[(b0 + tid) * Tlen + t + 1]) * (1.0f / 3.0f);
    __syncthreads();
    float hreg[8];
#pragma unroll
    for (int e = 0; e < 8; ++e) hreg[e] = hStep[e + hoff];

#pragma unroll 1
    for (int ss = 0; ss < 3; ++ss) {                  // RK4 substeps
      float kac[2][8];
#pragma unroll 1
      for (int ev = 0; ev < 4; ++ev) {                // k1..k4
        gemm_hid<8>(aPack, hPack, W1s, b1L, w, lane);
        __syncthreads();
        v8f kv[2];
        gemm_acc<16>(hPack, W2s, b2L, w, lane, kv);
        const float cw = (ev == 0 || ev == 3) ? 1.0f : 2.0f;
#pragma unroll
        for (int j = 0; j < 2; ++j)
#pragma unroll
          for (int e = 0; e < 8; ++e)
            kac[j][e] = (ev == 0) ? kv[j][e] : kac[j][e] + cw * kv[j][e];
        if (ev < 3) {
          const float cn = (ev == 2) ? 1.0f : 0.5f;
          float st[2][8];
#pragma unroll
          for (int j = 0; j < 2; ++j)
#pragma unroll
            for (int e = 0; e < 8; ++e)
              st[j][e] = s_own[j][e] + cn * hreg[e] * kv[j][e];
          pack_state(aPack, st, w, lane);
        } else {
#pragma unroll
          for (int j = 0; j < 2; ++j)
#pragma unroll
            for (int e = 0; e < 8; ++e)
              s_own[j][e] += (hreg[e] * (1.0f / 6.0f)) * kac[j][e];
          pack_state(aPack, s_own, w, lane);
        }
        __syncthreads();
      }
    }

    // RNN update with observation x_t
    pack_x(aPack, dataset, b0, t, tid);
    __syncthreads();
    gemm_hid<10>(aPack, hPack, Wr1s, br1L, w, lane);
    __syncthreads();
    v8f kv[2];
    gemm_acc<16>(hPack, Wr2s, br2L, w, lane, kv);
#pragma unroll
    for (int j = 0; j < 2; ++j)
#pragma unroll
      for (int e = 0; e < 8; ++e) s_own[j][e] += kv[j][e];
    pack_state(aPack, s_own, w, lane);
    __syncthreads();
  }

  // ---- write final state [B, LAT] f32 ----
#pragma unroll
  for (int j = 0; j < 2; ++j)
#pragma unroll
    for (int e = 0; e < 8; ++e)
      out[(b0 + e + hoff) * LAT + (w * 2 + j) * 16 + nl] = s_own[j][e];
}

extern "C" void kernel_launch(void* const* d_in, const int* in_sizes, int n_in,
                              void* d_out, int out_size, void* d_ws, size_t ws_size,
                              hipStream_t stream) {
  (void)in_sizes; (void)n_in; (void)out_size; (void)ws_size;
  const float* dataset = (const float*)d_in[0];
  const float* ts      = (const float*)d_in[1];
  const float* W1      = (const float*)d_in[2];
  const float* b1      = (const float*)d_in[3];
  const float* W2      = (const float*)d_in[4];
  const float* b2      = (const float*)d_in[5];
  const float* Wr1     = (const float*)d_in[6];
  const float* br1     = (const float*)d_in[7];
  const float* Wr2     = (const float*)d_in[8];
  const float* br2     = (const float*)d_in[9];

  char* ws = (char*)d_ws;
  unsigned short* W1s  = (unsigned short*)(ws + 0);         // 256*512*2 = 256 KB
  unsigned short* W2s  = (unsigned short*)(ws + 262144);    // 512*256*2 = 256 KB
  unsigned short* Wr1s = (unsigned short*)(ws + 524288);    // 320*512*2 = 320 KB
  unsigned short* Wr2s = (unsigned short*)(ws + 851968);    // 512*256*2 = 256 KB

  pack_weight<<<(LAT * HID + 255) / 256, 256, 0, stream>>>(W1, W1s, LAT, HID);
  pack_weight<<<(HID * LAT + 255) / 256, 256, 0, stream>>>(W2, W2s, HID, LAT);
  pack_weight<<<((LAT + OBS) * HID + 255) / 256, 256, 0, stream>>>(Wr1, Wr1s, LAT + OBS, HID);
  pack_weight<<<(HID * LAT + 255) / 256, 256, 0, stream>>>(Wr2, Wr2s, HID, LAT);

  odernn_main<<<Bsz / 16, 256, 0, stream>>>(dataset, ts, b1, b2, br1, br2,
                                            W1s, W2s, Wr1s, Wr2s, (float*)d_out);
}
